// EchoStateNetwork_24550033064324
// MI455X (gfx1250) — matrix operand order repaired and checked by reading, not compiled, not run on hardware
//
#include <hip/hip_runtime.h>
#include <hip/hip_bf16.h>

// ---------------- problem constants ----------------
#define RSIZE   2048
#define EMB     768
#define NFRAMES 256
#define NCLS    1000
#define NCLSP   1024          // classes padded to multiple of 16
#define BATCH   128
#define KTOT    (RSIZE + EMB) // fused K for recurrence GEMM: 2816
#define NKITER  (KTOT / 32)   // 88 K-iterations of 32

// LDS B tile: 64 rows (N) x 32 cols (K) bf16, row stride padded to 40 elems
// (80 bytes) -> 16-B aligned and bank-conflict-free for laneLo row indexing.
// Triple-buffered: 2 async tiles in flight, single barrier per iteration.
#define LDS_ROW  40
#define LDS_TILE (64 * LDS_ROW)   // elements per buffer
#define LDS_BUFS 3

typedef __bf16 bf16_t;
typedef bf16_t v16bf __attribute__((ext_vector_type(16)));
typedef bf16_t v8bf  __attribute__((ext_vector_type(8)));
typedef float  v8f   __attribute__((ext_vector_type(8)));

union V16 { v16bf v; v8bf h[2]; };

__device__ __forceinline__ v8f wmma_bf16(v16bf a, v16bf b, v8f c) {
  // D(f32 16x16) = A(16x32 bf16) x B(32x16 bf16) + C
  return __builtin_amdgcn_wmma_f32_16x16x32_bf16(
      /*neg_a=*/false, a, /*neg_b=*/false, b,
      /*c_mod=*/(short)0, c, /*reuse_a=*/false, /*reuse_b=*/false);
}

// Async copy 16 bytes/lane global -> LDS, tracked per-wave by ASYNCcnt.
__device__ __forceinline__ void async_g2l_b128(unsigned lds_byte_addr,
                                               const bf16_t* gptr) {
  asm volatile("global_load_async_to_lds_b128 %0, %1, off"
               :: "v"(lds_byte_addr), "v"(gptr)
               : "memory");
}
__device__ __forceinline__ void wait_async_le1() {
  asm volatile("s_wait_asynccnt 0x1" ::: "memory");
}
__device__ __forceinline__ void wait_async_0() {
  asm volatile("s_wait_asynccnt 0x0" ::: "memory");
}

// ---------------- conversion / layout kernels ----------------
__global__ void f32_to_bf16_kernel(const float* __restrict__ in,
                                   bf16_t* __restrict__ out, long n) {
  long i = (long)blockIdx.x * blockDim.x + threadIdx.x;
  long stride = (long)gridDim.x * blockDim.x;
  for (; i < n; i += stride) out[i] = (bf16_t)in[i];
}

// W_out [1000,2048] f32 -> [1024,2048] bf16, zero padded rows
__global__ void pad_wout_kernel(const float* __restrict__ in,
                                bf16_t* __restrict__ out) {
  long n = (long)NCLSP * RSIZE;
  long i = (long)blockIdx.x * blockDim.x + threadIdx.x;
  long stride = (long)gridDim.x * blockDim.x;
  for (; i < n; i += stride) {
    long row = i / RSIZE, col = i % RSIZE;
    out[i] = (row < NCLS) ? (bf16_t)in[row * RSIZE + col] : (bf16_t)0.0f;
  }
}

// x [B,T,E] f32 -> xT [T,B,E] bf16 (contiguous per frame)
__global__ void transpose_x_kernel(const float* __restrict__ x,
                                   bf16_t* __restrict__ xT) {
  long n = (long)NFRAMES * BATCH * EMB;
  long i = (long)blockIdx.x * blockDim.x + threadIdx.x;
  long stride = (long)gridDim.x * blockDim.x;
  for (; i < n; i += stride) {
    long e = i % EMB;
    long rest = i / EMB;
    long b = rest % BATCH;
    long t = rest / BATCH;
    xT[i] = (bf16_t)x[((size_t)b * NFRAMES + t) * EMB + e];
  }
}

// ---------------- recurrence step: Sout = tanh([Sin|Xt] @ [Wres|Win]^T) ----
// grid: RSIZE/64 blocks x 256 threads (8 waves). wave w -> M tile w*16,
// block b -> N tile b*64. B tile (64x32) triple-buffered in LDS via
// global_load_async_to_lds_b128, shared by all 8 waves (8x L2-traffic cut).
__global__ __launch_bounds__(256) void esn_step_kernel(
    const bf16_t* __restrict__ Sin,   // [BATCH, RSIZE]
    const bf16_t* __restrict__ Xt,    // [BATCH, EMB]   (frame t)
    const bf16_t* __restrict__ Wres,  // [RSIZE, RSIZE] row-major (N,K)
    const bf16_t* __restrict__ Win,   // [RSIZE, EMB]   row-major (N,K)
    bf16_t* __restrict__ Sout)        // [BATCH, RSIZE]
{
  __shared__ __align__(16) bf16_t lb[LDS_BUFS][LDS_TILE];

  const int lane   = threadIdx.x & 31;
  const int wave   = threadIdx.x >> 5;
  const int laneLo = lane & 15;
  const int laneHi = lane >> 4;

  const int m0 = wave * 16;
  const int n0 = blockIdx.x * 64;
  const int mRow = m0 + laneLo;

  // stage mapping: 256 threads x 16B = 4KB tile. row = N index, chunk = 8-elem K slice
  const int srow   = threadIdx.x >> 2;
  const int schunk = threadIdx.x & 3;
  // LDS byte address: flat shared pointer truncated to 32 bits == LDS address
  const unsigned ldsDst0 =
      (unsigned)(uintptr_t)(&lb[0][srow * LDS_ROW + schunk * 8]);
  const unsigned bufBytes = (unsigned)(LDS_TILE * sizeof(bf16_t));

  v8f acc[4] = {};

  // prologue: stage K tiles 0 and 1 into buffers 0 and 1 (2 in flight)
  {
    const bf16_t* g0 = Wres + (size_t)(n0 + srow) * RSIZE + schunk * 8;
    async_g2l_b128(ldsDst0, g0);
    const bf16_t* g1 = Wres + (size_t)(n0 + srow) * RSIZE + 32 + schunk * 8;
    async_g2l_b128(ldsDst0 + bufBytes, g1);
  }

  unsigned rdOff = 0;                 // byte offset of buffer being read
  unsigned wrOff = 2u * bufBytes;     // byte offset of buffer being filled next
  for (int it = 0; it < NKITER; ++it) {
    const int k = it * 32;

    // ---- A tile (16x32 bf16), direct from global; issue early so its
    // latency hides under the async-wait + barrier.
    // lanes 0-15 hold K {0-7,16-23}; lanes 16-31 hold K {8-15,24-31}.
    V16 a;
    if (k < RSIZE) {
      const bf16_t* arow = Sin + (size_t)mRow * RSIZE + k + laneHi * 8;
      a.h[0] = *(const v8bf*)(arow);
      a.h[1] = *(const v8bf*)(arow + 16);
    } else {
      const bf16_t* arow = Xt + (size_t)mRow * EMB + (k - RSIZE) + laneHi * 8;
      a.h[0] = *(const v8bf*)(arow);
      a.h[1] = *(const v8bf*)(arow + 16);
    }

    // retire stage(it); leave stage(it+1) in flight
    if (it + 1 < NKITER) wait_async_le1(); else wait_async_0();
    __syncthreads();   // buf(it) visible everywhere; all reads of buf(it-1) done

    // issue stage(it+2) into the buffer read at it-1 (safe: post-barrier)
    if (it + 2 < NKITER) {
      const int kn = k + 64;
      const bf16_t* g;
      if (kn < RSIZE)
        g = Wres + (size_t)(n0 + srow) * RSIZE + kn + schunk * 8;
      else
        g = Win + (size_t)(n0 + srow) * EMB + (kn - RSIZE) + schunk * 8;
      async_g2l_b128(ldsDst0 + wrOff, g);
    }

    // ---- B tiles from LDS: load all four, then run the WMMAs back-to-back.
    const bf16_t* lbuf = (const bf16_t*)((const char*)&lb[0][0] + rdOff);
    V16 b[4];
#pragma unroll
    for (int c = 0; c < 4; ++c) {
      const bf16_t* p = lbuf + (c * 16 + laneLo) * LDS_ROW + laneHi * 8;
      b[c].h[0] = *(const v8bf*)(p);
      b[c].h[1] = *(const v8bf*)(p + 16);
    }
#pragma unroll
    for (int c = 0; c < 4; ++c)
      acc[c] = wmma_bf16(a.v, b[c].v, acc[c]);

    // rotate triple buffer
    rdOff = (rdOff == 2u * bufBytes) ? 0u : rdOff + bufBytes;
    wrOff = (wrOff == 2u * bufBytes) ? 0u : wrOff + bufBytes;
  }

  // ---- epilogue: tanh, store bf16 state.
  // D layout: VGPR r -> M = m0 + r + 8*laneHi, N = n0 + 16*c + laneLo.
#pragma unroll
  for (int c = 0; c < 4; ++c) {
    const int n = n0 + c * 16 + laneLo;
#pragma unroll
    for (int r = 0; r < 8; ++r) {
      const int m = m0 + r + laneHi * 8;
      Sout[(size_t)m * RSIZE + n] = (bf16_t)tanhf(acc[c][r]);
    }
  }
}

// ---------------- head: logits = S @ Wout^T + b ---------------------------
__global__ __launch_bounds__(256) void esn_head_kernel(
    const bf16_t* __restrict__ S,     // [BATCH, RSIZE]
    const bf16_t* __restrict__ Wout,  // [NCLSP, RSIZE]
    const float*  __restrict__ bout,  // [NCLS]
    float* __restrict__ out)          // [BATCH, NCLS]
{
  const int lane   = threadIdx.x & 31;
  const int wave   = threadIdx.x >> 5;
  const int laneLo = lane & 15;
  const int laneHi = lane >> 4;

  const int m0 = wave * 16;
  const int n0 = blockIdx.x * 64;
  const int mRow = m0 + laneLo;

  v8f acc[4] = {};

  for (int k = 0; k < RSIZE; k += 32) {
    V16 a;
    const bf16_t* arow = S + (size_t)mRow * RSIZE + k + laneHi * 8;
    a.h[0] = *(const v8bf*)(arow);
    a.h[1] = *(const v8bf*)(arow + 16);
#pragma unroll
    for (int c = 0; c < 4; ++c) {
      const int n = n0 + c * 16 + laneLo;
      V16 b;
      const bf16_t* brow = Wout + (size_t)n * RSIZE + k + laneHi * 8;
      b.h[0] = *(const v8bf*)(brow);
      b.h[1] = *(const v8bf*)(brow + 16);
      acc[c] = wmma_bf16(a.v, b.v, acc[c]);
    }
  }

#pragma unroll
  for (int c = 0; c < 4; ++c) {
    const int n = n0 + c * 16 + laneLo;
    if (n < NCLS) {
      const float bias = bout[n];
#pragma unroll
      for (int r = 0; r < 8; ++r) {
        const int m = m0 + r + laneHi * 8;
        out[(size_t)m * NCLS + n] = acc[c][r] + bias;
      }
    }
  }
}

// ---------------- host orchestration --------------------------------------
extern "C" void kernel_launch(void* const* d_in, const int* in_sizes, int n_in,
                              void* d_out, int out_size, void* d_ws, size_t ws_size,
                              hipStream_t stream) {
  const float* x         = (const float*)d_in[0]; // [B,T,E]
  const float* reservoir = (const float*)d_in[1]; // [R,R]
  const float* W_in      = (const float*)d_in[2]; // [R,E]
  const float* W_out     = (const float*)d_in[3]; // [C,R]
  const float* b_out     = (const float*)d_in[4]; // [C]
  float* out = (float*)d_out;

  char* ws = (char*)d_ws;
  bf16_t* Wres = (bf16_t*)ws; ws += (size_t)RSIZE * RSIZE * sizeof(bf16_t);
  bf16_t* Win  = (bf16_t*)ws; ws += (size_t)RSIZE * EMB   * sizeof(bf16_t);
  bf16_t* Wout = (bf16_t*)ws; ws += (size_t)NCLSP * RSIZE * sizeof(bf16_t);
  bf16_t* xT   = (bf16_t*)ws; ws += (size_t)NFRAMES * BATCH * EMB * sizeof(bf16_t);
  bf16_t* s0   = (bf16_t*)ws; ws += (size_t)BATCH * RSIZE * sizeof(bf16_t);
  bf16_t* s1   = (bf16_t*)ws;

  // state(t=0) = 0
  hipMemsetAsync(s0, 0, (size_t)BATCH * RSIZE * sizeof(bf16_t), stream);

  // one-time precision/layout conversion (fully parallel, HBM-bound)
  {
    long nres = (long)RSIZE * RSIZE;
    f32_to_bf16_kernel<<<2048, 256, 0, stream>>>(reservoir, Wres, nres);
    long nwin = (long)RSIZE * EMB;
    f32_to_bf16_kernel<<<2048, 256, 0, stream>>>(W_in, Win, nwin);
    pad_wout_kernel<<<2048, 256, 0, stream>>>(W_out, Wout);
    transpose_x_kernel<<<4096, 256, 0, stream>>>(x, xT);
  }

  // 256 serial recurrence steps (weights stay L2-resident)
  bf16_t* sin  = s0;
  bf16_t* sout = s1;
  for (int t = 0; t < NFRAMES; ++t) {
    esn_step_kernel<<<dim3(RSIZE / 64), 256, 0, stream>>>(
        sin, xT + (size_t)t * BATCH * EMB, Wres, Win, sout);
    bf16_t* tmp = sin; sin = sout; sout = tmp;
  }

  // linear head on final state (held in `sin` after the last swap)
  esn_head_kernel<<<dim3(NCLSP / 64), 256, 0, stream>>>(sin, Wout, b_out, out);
}
